// TransformerHATLayer_2576980378139
// MI455X (gfx1250) — matrix operand order repaired and checked by reading, not compiled, not run on hardware
//
#include <hip/hip_runtime.h>
#include <stdint.h>

typedef __bf16 bf16;
typedef __attribute__((ext_vector_type(8)))  bf16  v8bf;
typedef __attribute__((ext_vector_type(16))) bf16  v16bf;
typedef __attribute__((ext_vector_type(8)))  float v8f;
typedef __attribute__((ext_vector_type(4)))  unsigned int u32x4;
typedef __attribute__((ext_vector_type(8)))  int   i32x8;
typedef __attribute__((ext_vector_type(4)))  int   i32x4;

#define DEV __device__ __forceinline__

// ---------------- problem sizes ----------------
constexpr int N_NODES = 65536;
constexpr int N_EDGES = 32768;

// ---------------- parameter offsets (floats) ----------------
// jax tree_flatten order: alphabetical at each dict level.
constexpr int SZ_W = 128 * 128, SZ_B = 128, SZ_I = 4 * 128, SZ_WW = 16 * 128;
constexpr int DEC_W  = 0;
constexpr int DEC_B  = DEC_W + SZ_W;
constexpr int IMAB_I = DEC_B + SZ_B;
constexpr int M0_WW  = IMAB_I + SZ_I;
constexpr int M0_KW  = M0_WW + SZ_WW;
constexpr int M0_KB  = M0_KW + SZ_W;
constexpr int M0_OW  = M0_KB + SZ_B;
constexpr int M0_OB  = M0_OW + SZ_W;
constexpr int M0_QW  = M0_OB + SZ_B;
constexpr int M0_QB  = M0_QW + SZ_W;
constexpr int M0_VW  = M0_QB + SZ_B;
constexpr int M0_VB  = M0_VW + SZ_W;
constexpr int M1_KW  = M0_VB + SZ_B;
constexpr int M1_KB  = M1_KW + SZ_W;
constexpr int M1_OW  = M1_KB + SZ_B;
constexpr int M1_OB  = M1_OW + SZ_W;
constexpr int M1_QW  = M1_OB + SZ_B;
constexpr int M1_QB  = M1_QW + SZ_W;
constexpr int M1_VW  = M1_QB + SZ_B;
constexpr int M1_VB  = M1_VW + SZ_W;
constexpr int ISAB_I = M1_VB + SZ_B;
constexpr int S0_KW  = ISAB_I + SZ_I;
constexpr int S0_KB  = S0_KW + SZ_W;
constexpr int S0_OW  = S0_KB + SZ_B;
constexpr int S0_OB  = S0_OW + SZ_W;
constexpr int S0_QW  = S0_OB + SZ_B;
constexpr int S0_QB  = S0_QW + SZ_W;
constexpr int S0_VW  = S0_QB + SZ_B;
constexpr int S0_VB  = S0_VW + SZ_W;
constexpr int S1_KW  = S0_VB + SZ_B;
constexpr int S1_KB  = S1_KW + SZ_W;
constexpr int S1_OW  = S1_KB + SZ_B;
constexpr int S1_OB  = S1_OW + SZ_W;
constexpr int S1_QW  = S1_OB + SZ_B;
constexpr int S1_QB  = S1_QW + SZ_W;
constexpr int S1_VW  = S1_QB + SZ_B;
constexpr int S1_VB  = S1_VW + SZ_W;
constexpr int KE_W   = S1_VB + SZ_B;
constexpr int KE_B   = KE_W + SZ_W;
constexpr int QV_W   = KE_B + SZ_B;
constexpr int QV_B   = QV_W + SZ_W;
constexpr int VE_W   = QV_B + SZ_B;
constexpr int VE_B   = VE_W + SZ_W;
constexpr int WT_OFF = VE_B + SZ_B;   // wt [16,128]

// ---------------- workspace layout (bytes) ----------------
constexpr size_t WS_PACKW = 0;                                   // 19 matrices, WMMA B-frag order, bf16
constexpr size_t WS_WW    = WS_PACKW + (size_t)19 * 16384 * 2;   // Ww packed (K padded to 32), 4096 bf16
constexpr size_t WS_IPAD  = WS_WW    + 4096 * 2;                 // 2 x [16][128] bf16 (I padded)
constexpr size_t WS_VFB   = WS_IPAD  + 2 * 2048 * 2;             // vfeat bf16
constexpr size_t WS_EFB   = WS_VFB   + (size_t)N_NODES * 128 * 2;// efeat_new bf16
constexpr size_t WS_Q     = WS_EFB   + (size_t)N_EDGES * 128 * 2;// q f32
constexpr size_t WS_K     = WS_Q     + (size_t)N_NODES * 128 * 4;// k f32
constexpr size_t WS_VE    = WS_K     + (size_t)N_EDGES * 128 * 4;// ve f32

// ---------------- stage-1 LDS layout ----------------
constexpr int LDS_WSTAGE = 0;                 // 4 matrices * 32768 B (TDM destination)
constexpr int LDS_EDGE0  = 4 * 32768;         // per-wave (per-edge) regions
constexpr int PE_XV  = 0;      // [16][128] bf16
constexpr int PE_HB  = 4096;   // [16][128] bf16
constexpr int PE_QP  = 8192;
constexpr int PE_KP  = 12288;
constexpr int PE_VP  = 16384;
constexpr int PE_VPT = 20480;  // [128][16] bf16 (V transposed)
constexpr int PE_OB  = 24576;
constexpr int PE_W1P = 28672;  // [16][32] bf16
constexpr int PE_AB  = 29696;  // [16][32] bf16 (softmax probs)
constexpr int PE_SB  = 30720;  // [16][16] f32 scores / scratch
constexpr int PE_SZ  = 31744;
constexpr int LDS_TOTAL = LDS_EDGE0 + 4 * PE_SZ;  // 258048 B < 320 KB/WGP

// ---------------- helpers ----------------
DEV void lds_fence() { asm volatile("s_wait_dscnt 0" ::: "memory"); }

DEV v16bf catbf(v8bf lo, v8bf hi) {
  return __builtin_shufflevector(lo, hi, 0,1,2,3,4,5,6,7,8,9,10,11,12,13,14,15);
}
DEV v16bf load16(const bf16* p) {                    // 16 contiguous bf16
  return catbf(*(const v8bf*)p, *(const v8bf*)(p + 8));
}
// A-fragment: [16][stride] matrix, rows=M, K-chunk base kc (contiguous k = 16*(lane/16) + s)
DEV v16bf loadA(const bf16* buf, int stride, int kc, int lane) {
  int m = lane & 15, kb = (lane >> 4) * 16;
  const bf16* p = buf + m * stride + kc + kb;
  return load16(p);
}
// B-fragment from packed weights: layout [kt][nt][lane][16]
DEV v16bf loadBpk(const bf16* Wp, int kt, int nt, int lane) {
  return load16(Wp + (((kt << 3) + nt) * 32 + lane) * 16);
}
DEV v8f wmma_bf(v16bf a, v16bf b, v8f c) {
  return __builtin_amdgcn_wmma_f32_16x16x32_bf16(false, a, false, b, (short)0, c, false, false);
}
DEV v16bf zerobf() {
  v16bf z;
  #pragma unroll
  for (int i = 0; i < 16; ++i) z[i] = (bf16)0.0f;
  return z;
}
DEV v8f zerof() {
  v8f z;
  #pragma unroll
  for (int i = 0; i < 8; ++i) z[i] = 0.0f;
  return z;
}

// TDM: DMA 32 KB (4096 x u64) from global into LDS at byte offset lds_off.
DEV void tdm_load32k(const bf16* gsrc, unsigned lds_off) {
  unsigned long long ga = (unsigned long long)(uintptr_t)gsrc;
  u32x4 g0;
  g0[0] = 1u;                                              // count=1, user mode
  g0[1] = lds_off;                                         // lds_addr
  g0[2] = (unsigned)(ga & 0xffffffffu);                    // global_addr lo
  g0[3] = (unsigned)((ga >> 32) & 0x1ffffffu) | (2u << 30);// global_addr hi | type=2
  i32x8 g1;
  g1[0] = (3 << 16);              // data_size = 8B
  g1[1] = (int)(4096u << 16);     // tensor_dim0 = 4096 (lo16)
  g1[2] = (int)(1u << 16);        // tensor_dim0 hi=0 | tensor_dim1 = 1
  g1[3] = (int)(4096u << 16);     // tile_dim0 = 4096
  g1[4] = 0;                      // tile_dim1/2 unused
  g1[5] = 4096;                   // tensor_dim0_stride lo32
  g1[6] = 0;
  g1[7] = 0;
  i32x4 z4;
  z4[0] = z4[1] = z4[2] = z4[3] = 0;
  i32x8 z8;
  #pragma unroll
  for (int i = 0; i < 8; ++i) z8[i] = 0;
  __builtin_amdgcn_tensor_load_to_lds(g0, g1, z4, z4, z8, 0);
}

// out[16][128] = A[16][128] @ W + bias  (+ optional extra K=16 term, +relu-residual, +transposed copy)
DEV void gemm16(bf16* out, const bf16* A, const bf16* Wp, const float* bias,
                const bf16* resid, bf16* outT, const bf16* xA, const bf16* xB, int lane) {
  int n = lane & 15, hi = lane >> 4;
  v8f acc[8];
  #pragma unroll
  for (int nt = 0; nt < 8; ++nt) {
    float bv = bias[nt * 16 + n];
    #pragma unroll
    for (int i = 0; i < 8; ++i) acc[nt][i] = bv;
  }
  #pragma unroll
  for (int kt = 0; kt < 4; ++kt) {
    v16bf a = loadA(A, 128, kt * 32, lane);
    #pragma unroll
    for (int nt = 0; nt < 8; ++nt) acc[nt] = wmma_bf(a, loadBpk(Wp, kt, nt, lane), acc[nt]);
  }
  if (xA) {  // w1 @ Ww term (K = 16 padded to 32)
    v16bf a = loadA(xA, 32, 0, lane);
    #pragma unroll
    for (int nt = 0; nt < 8; ++nt) acc[nt] = wmma_bf(a, loadBpk(xB, 0, nt, lane), acc[nt]);
  }
  #pragma unroll
  for (int nt = 0; nt < 8; ++nt) {
    #pragma unroll
    for (int r = 0; r < 8; ++r) {
      int row = r + hi * 8, col = nt * 16 + n;
      float v = acc[nt][r];
      if (resid) v = (float)resid[row * 128 + col] + fmaxf(v, 0.0f);  // O + relu(fc_o(O))
      out[row * 128 + col] = (bf16)v;
      if (outT) outT[col * 16 + row] = (bf16)v;
    }
  }
  lds_fence();
}

// multi-head attention (4 heads x 32): Ob = Qp + softmax(Qp Kp^T / sqrt(128)) Vp
DEV void attention(bf16* Ob, const bf16* Qp, const bf16* Kp, const bf16* VpT,
                   float* Sb, bf16* Ab, int Lk, int lane) {
  const float scale = 0.088388347648318447f;  // 1/sqrt(128)
  int n = lane & 15, hi = lane >> 4;
  #pragma unroll
  for (int h = 0; h < 4; ++h) {
    // scores: A = Qp_h [16x32], B[k][n] = Kp[n][h*32+k] (contiguous in Kp rows)
    v16bf a = loadA(Qp, 128, h * 32, lane);
    v16bf b = load16(Kp + n * 128 + h * 32 + hi * 16);
    v8f s = zerof();
    s = wmma_bf(a, b, s);
    #pragma unroll
    for (int r = 0; r < 8; ++r) Sb[(r + hi * 8) * 16 + n] = s[r] * scale;
    lds_fence();
    // row softmax over valid keys (mask Lk)
    if (lane < 16) {
      float sv[16], mx = -1e30f;
      for (int k = 0; k < Lk; ++k) { sv[k] = Sb[lane * 16 + k]; mx = fmaxf(mx, sv[k]); }
      float sum = 0.0f;
      for (int k = 0; k < Lk; ++k) { sv[k] = __expf(sv[k] - mx); sum += sv[k]; }
      float inv = 1.0f / sum;
      for (int k = 0; k < 16; ++k) Ab[lane * 32 + k] = (k < Lk) ? (bf16)(sv[k] * inv) : (bf16)0.0f;
      for (int k = 16; k < 32; ++k) Ab[lane * 32 + k] = (bf16)0.0f;
    }
    lds_fence();
    // O_h = A @ Vp_h via transposed V copy; residual add Qp
    v16bf av = loadA(Ab, 32, 0, lane);
    #pragma unroll
    for (int t = 0; t < 2; ++t) {
      int c = h * 32 + t * 16 + n;
      v16bf bv = (hi == 0) ? load16(VpT + c * 16) : zerobf();
      v8f o = zerof();
      o = wmma_bf(av, bv, o);
      #pragma unroll
      for (int r = 0; r < 8; ++r) {
        int row = r + hi * 8;
        Ob[row * 128 + c] = (bf16)((float)Qp[row * 128 + c] + o[r]);
      }
    }
    lds_fence();
  }
}

// ---------------- pack kernels ----------------
__device__ const int SRC_OFF[19] = {
  M0_QW, M0_KW, M0_VW, M0_OW,  M1_QW, M1_KW, M1_VW, M1_OW,
  S0_QW, S0_KW, S0_VW, S0_OW,  S1_QW, S1_KW, S1_VW, S1_OW,
  QV_W, KE_W, VE_W };

__global__ void __launch_bounds__(256) pack_weights_kernel(const float* __restrict__ P,
                                                           bf16* __restrict__ dst) {
  int d = blockIdx.x * 256 + threadIdx.x;
  if (d >= 19 * 16384) return;
  int mi = d >> 14, r = d & 16383;
  int kt = r >> 12, r2 = r & 4095;
  int nt = r2 >> 9, r3 = r2 & 511;
  int lane = r3 >> 4, el = r3 & 15;
  int k = kt * 32 + ((lane >> 4) << 4) + el;
  int n = nt * 16 + (lane & 15);
  dst[d] = (bf16)P[SRC_OFF[mi] + k * 128 + n];
}

__global__ void __launch_bounds__(256) pack_ww_kernel(const float* __restrict__ P,
                                                      bf16* __restrict__ dst) {
  int d = blockIdx.x * 256 + threadIdx.x;
  if (d >= 4096) return;
  int nt = d >> 9, r3 = d & 511;
  int lane = r3 >> 4, el = r3 & 15;
  int kl = ((lane >> 4) << 4) + el;
  int n = nt * 16 + (lane & 15);
  dst[d] = (kl < 16) ? (bf16)P[M0_WW + kl * 128 + n] : (bf16)0.0f;
}

__global__ void __launch_bounds__(256) pack_ipad_kernel(const float* __restrict__ P,
                                                        bf16* __restrict__ dst) {
  int d = blockIdx.x * 256 + threadIdx.x;
  if (d >= 4096) return;
  int b = d >> 11, r = d & 2047, row = r >> 7, col = r & 127;
  dst[d] = (row < 4) ? (bf16)P[(b ? ISAB_I : IMAB_I) + row * 128 + col] : (bf16)0.0f;
}

__global__ void __launch_bounds__(256) cvt_bf16_kernel(const float* __restrict__ src,
                                                       bf16* __restrict__ dst, int n) {
  int i = blockIdx.x * 256 + threadIdx.x;
  if (i < n) dst[i] = (bf16)src[i];
}

// ---------------- stage 1: per-edge set transformer ----------------
__device__ const int QB_OFF[4] = { M0_QB, M1_QB, S0_QB, S1_QB };
__device__ const int KB_OFF[4] = { M0_KB, M1_KB, S0_KB, S1_KB };
__device__ const int VB_OFF[4] = { M0_VB, M1_VB, S0_VB, S1_VB };
__device__ const int OB_OFF[4] = { M0_OB, M1_OB, S0_OB, S1_OB };

__global__ void __launch_bounds__(128)
stage1_kernel(const bf16* __restrict__ vfb, const float* __restrict__ w1,
              const int* __restrict__ nn, const float* __restrict__ P,
              const bf16* __restrict__ packW, const bf16* __restrict__ wwP,
              const bf16* __restrict__ ipad, float* __restrict__ efOut,
              bf16* __restrict__ efb) {
  extern __shared__ char smem[];
  const int tid = threadIdx.x, lane = tid & 31, wv = tid >> 5;
  const int e = blockIdx.x * 4 + wv;

  char* pe = smem + LDS_EDGE0 + wv * PE_SZ;
  bf16* Xv  = (bf16*)(pe + PE_XV);
  bf16* Hb  = (bf16*)(pe + PE_HB);
  bf16* Qp  = (bf16*)(pe + PE_QP);
  bf16* Kp  = (bf16*)(pe + PE_KP);
  bf16* Vp  = (bf16*)(pe + PE_VP);
  bf16* VpT = (bf16*)(pe + PE_VPT);
  bf16* Ob  = (bf16*)(pe + PE_OB);
  bf16* W1p = (bf16*)(pe + PE_W1P);
  bf16* Ab  = (bf16*)(pe + PE_AB);
  float* Sb = (float*)(pe + PE_SB);

  // gather node features [16][128] bf16 (8 B per lane per row)
  for (int r = 0; r < 16; ++r) {
    int node = nn[e * 16 + r];
    *(uint64_t*)(Xv + r * 128 + lane * 4) =
        *(const uint64_t*)(vfb + (size_t)node * 128 + lane * 4);
  }
  // incidence weights, K padded 16->32
  if (lane < 16) {
    for (int j = 0; j < 16; ++j) W1p[lane * 32 + j] = (bf16)w1[((size_t)e * 16 + lane) * 16 + j];
    for (int j = 16; j < 32; ++j) W1p[lane * 32 + j] = (bf16)0.0f;
  }
  lds_fence();

  for (int s = 0; s < 4; ++s) {
    // each wave TDM-loads one of the 4 stage matrices (q,k,v,o) into LDS
    tdm_load32k(packW + (size_t)(s * 4 + wv) * 16384, (unsigned)(LDS_WSTAGE + wv * 32768));
    __builtin_amdgcn_s_wait_tensorcnt(0);
    __syncthreads();
    const bf16* Wq = (const bf16*)(smem + LDS_WSTAGE + 0 * 32768);
    const bf16* Wk = (const bf16*)(smem + LDS_WSTAGE + 1 * 32768);
    const bf16* Wv = (const bf16*)(smem + LDS_WSTAGE + 2 * 32768);
    const bf16* Wo = (const bf16*)(smem + LDS_WSTAGE + 3 * 32768);

    bf16 *Qin, *Kin, *dst;
    int Lk;
    if ((s & 1) == 0) {             // mab0: Q = inducing points, K = Xv
      const bf16* ip = ipad + (s == 0 ? 0 : 2048);
      for (int r = 0; r < 16; ++r)
        *(uint64_t*)(Hb + r * 128 + lane * 4) = *(const uint64_t*)(ip + r * 128 + lane * 4);
      lds_fence();
      Qin = Hb; Kin = Xv; Lk = 16; dst = Hb;
    } else {                        // mab1: Q = Xv, K = H (4 valid rows)
      Qin = Xv; Kin = Hb; Lk = 4; dst = Xv;
    }

    gemm16(Qp, Qin, Wq, P + QB_OFF[s], nullptr, nullptr, nullptr, nullptr, lane);
    gemm16(Kp, Kin, Wk, P + KB_OFF[s], nullptr, nullptr,
           (s == 0) ? W1p : nullptr, (s == 0) ? wwP : nullptr, lane);
    gemm16(Vp, Kin, Wv, P + VB_OFF[s], nullptr, VpT, nullptr, nullptr, lane);
    attention(Ob, Qp, Kp, VpT, Sb, Ab, Lk, lane);
    gemm16(dst, Ob, Wo, P + OB_OFF[s], Ob, nullptr, nullptr, nullptr, lane);
    __syncthreads();
  }

  // mean-pool over 16 rows + decoder (1x128 @ 128x128)
  int c0 = lane * 4;
  float o0 = 0, o1 = 0, o2 = 0, o3 = 0;
  for (int m = 0; m < 16; ++m) {
    const bf16* row = Xv + m * 128 + c0;
    o0 += (float)row[0]; o1 += (float)row[1]; o2 += (float)row[2]; o3 += (float)row[3];
  }
  float* oL = Sb;  // 128 floats scratch
  oL[c0 + 0] = o0 * 0.0625f; oL[c0 + 1] = o1 * 0.0625f;
  oL[c0 + 2] = o2 * 0.0625f; oL[c0 + 3] = o3 * 0.0625f;
  lds_fence();
  const float* dW = P + DEC_W;
  const float* dB = P + DEC_B;
  float f0 = dB[c0 + 0], f1 = dB[c0 + 1], f2 = dB[c0 + 2], f3 = dB[c0 + 3];
  for (int k = 0; k < 128; ++k) {
    float ok = oL[k];
    const float* wr = dW + k * 128 + c0;
    f0 += ok * wr[0]; f1 += ok * wr[1]; f2 += ok * wr[2]; f3 += ok * wr[3];
  }
  size_t base = (size_t)e * 128 + c0;
  efOut[base + 0] = f0; efOut[base + 1] = f1; efOut[base + 2] = f2; efOut[base + 3] = f3;
  efb[base + 0] = (bf16)f0; efb[base + 1] = (bf16)f1;
  efb[base + 2] = (bf16)f2; efb[base + 3] = (bf16)f3;
}

// ---------------- generic row-block GEMM: out[M,128] = A[M,128] @ W + b ----------------
__global__ void __launch_bounds__(256)
gemm_rows_kernel(const bf16* __restrict__ A, const bf16* __restrict__ Wp,
                 const float* __restrict__ bias, float* __restrict__ out, int M) {
  int lane = threadIdx.x & 31, wv = threadIdx.x >> 5;
  int rowBase = (blockIdx.x * 8 + wv) * 16;
  if (rowBase >= M) return;
  int n = lane & 15, hi = lane >> 4, kb = hi * 16, m = lane & 15;
  v8f acc[8];
  #pragma unroll
  for (int nt = 0; nt < 8; ++nt) {
    float bv = bias[nt * 16 + n];
    #pragma unroll
    for (int i = 0; i < 8; ++i) acc[nt][i] = bv;
  }
  #pragma unroll
  for (int kt = 0; kt < 4; ++kt) {
    const bf16* p = A + (size_t)(rowBase + m) * 128 + kt * 32 + kb;
    v16bf a = catbf(*(const v8bf*)p, *(const v8bf*)(p + 8));
    #pragma unroll
    for (int nt = 0; nt < 8; ++nt) acc[nt] = wmma_bf(a, loadBpk(Wp, kt, nt, lane), acc[nt]);
  }
  #pragma unroll
  for (int nt = 0; nt < 8; ++nt) {
    #pragma unroll
    for (int r = 0; r < 8; ++r)
      out[(size_t)(rowBase + r + hi * 8) * 128 + nt * 16 + n] = acc[nt][r];
  }
}

// ---------------- stage 2: hyperedge -> node attention scatter ----------------
__global__ void __launch_bounds__(256)
stage2_kernel(const float* __restrict__ qbuf, const float* __restrict__ kbuf,
              const float* __restrict__ vebuf, const int* __restrict__ ne,
              const float* __restrict__ w2, const float* __restrict__ P,
              float* __restrict__ out) {
  const float scale = 0.088388347648318447f;  // 1/sqrt(128)
  int lane = threadIdx.x & 31, wv = threadIdx.x >> 5;
  int node = blockIdx.x * 8 + wv;
  int c = lane * 4;
  const float* wt = P + WT_OFF;
  const float* qr = qbuf + (size_t)node * 128 + c;
  float q0 = qr[0], q1 = qr[1], q2 = qr[2], q3 = qr[3];
  float sc[8];
  int eidx[8];
  for (int d = 0; d < 8; ++d) {
    int ed = ne[node * 8 + d];
    eidx[d] = ed;
    const float* kr = kbuf + (size_t)ed * 128 + c;
    float k0 = kr[0], k1 = kr[1], k2 = kr[2], k3 = kr[3];
    const float* w2p = w2 + ((size_t)node * 8 + d) * 16;
    for (int j = 0; j < 16; ++j) {
      float wj = w2p[j];
      const float* wr = wt + j * 128 + c;
      k0 += wj * wr[0]; k1 += wj * wr[1]; k2 += wj * wr[2]; k3 += wj * wr[3];
    }
    float p = q0 * k0 + q1 * k1 + q2 * k2 + q3 * k3;
    for (int off = 16; off; off >>= 1) p += __shfl_xor(p, off, 32);
    sc[d] = (p < 0.0f ? 0.01f * p : p) * scale;  // leaky_relu then scale
  }
  float mx = sc[0];
  for (int d = 1; d < 8; ++d) mx = fmaxf(mx, sc[d]);
  float sum = 0.0f;
  for (int d = 0; d < 8; ++d) { sc[d] = __expf(sc[d] - mx); sum += sc[d]; }
  float inv = 1.0f / sum;
  float h0 = 0, h1 = 0, h2 = 0, h3 = 0;
  for (int d = 0; d < 8; ++d) {
    float a = sc[d] * inv;
    const float* vr = vebuf + (size_t)eidx[d] * 128 + c;
    h0 += a * vr[0]; h1 += a * vr[1]; h2 += a * vr[2]; h3 += a * vr[3];
  }
  size_t b = (size_t)node * 128 + c;
  out[b + 0] = fmaxf(h0, 0.0f); out[b + 1] = fmaxf(h1, 0.0f);
  out[b + 2] = fmaxf(h2, 0.0f); out[b + 3] = fmaxf(h3, 0.0f);
}

// ---------------- launch ----------------
extern "C" void kernel_launch(void* const* d_in, const int* in_sizes, int n_in,
                              void* d_out, int out_size, void* d_ws, size_t ws_size,
                              hipStream_t stream) {
  const float* vfeat = (const float*)d_in[0];
  // d_in[1] = efeat (unused by the reference)
  const float* w1 = (const float*)d_in[2];
  const float* w2 = (const float*)d_in[3];
  const float* P  = (const float*)d_in[4];
  const int* nn   = (const int*)d_in[5];
  const int* ne   = (const int*)d_in[6];
  float* out = (float*)d_out;

  char* ws = (char*)d_ws;
  bf16* packW = (bf16*)(ws + WS_PACKW);
  bf16* wwP   = (bf16*)(ws + WS_WW);
  bf16* ipad  = (bf16*)(ws + WS_IPAD);
  bf16* vfb   = (bf16*)(ws + WS_VFB);
  bf16* efb   = (bf16*)(ws + WS_EFB);
  float* qbuf = (float*)(ws + WS_Q);
  float* kbuf = (float*)(ws + WS_K);
  float* vebuf = (float*)(ws + WS_VE);

  pack_weights_kernel<<<(19 * 16384 + 255) / 256, 256, 0, stream>>>(P, packW);
  pack_ww_kernel<<<16, 256, 0, stream>>>(P, wwP);
  pack_ipad_kernel<<<16, 256, 0, stream>>>(P, ipad);
  cvt_bf16_kernel<<<(N_NODES * 128) / 256, 256, 0, stream>>>(vfeat, vfb, N_NODES * 128);

  stage1_kernel<<<N_EDGES / 4, 128, LDS_TOTAL, stream>>>(
      vfb, w1, nn, P, packW, wwP, ipad, out + (size_t)N_NODES * 128, efb);

  gemm_rows_kernel<<<N_NODES / 128, 256, 0, stream>>>(vfb, packW + 16 * 16384, P + QV_B, qbuf, N_NODES);
  gemm_rows_kernel<<<N_EDGES / 128, 256, 0, stream>>>(efb, packW + 17 * 16384, P + KE_B, kbuf, N_EDGES);
  gemm_rows_kernel<<<N_EDGES / 128, 256, 0, stream>>>(efb, packW + 18 * 16384, P + VE_B, vebuf, N_EDGES);

  stage2_kernel<<<N_NODES / 8, 256, 0, stream>>>(qbuf, kbuf, vebuf, ne, w2, P, out);
}
